// SetAbstraction_79422535238105
// MI455X (gfx1250) — matrix operand order repaired and checked by reading, not compiled, not run on hardware
//
#include <hip/hip_runtime.h>
#include <hip/hip_bf16.h>
#include <math.h>

#define PB 8
#define PN 6000
#define PNPAD 6016
#define PCIN 128
#define PCOUT 256
#define PH 4
#define PNP 1500
#define PNPPAD 1536
#define PCH 64
#define PM 150
#define PMPAD 192
#define PR2 0.04f

typedef __attribute__((ext_vector_type(16))) _Float16 v16h;
typedef __attribute__((ext_vector_type(8)))  _Float16 v8h;
typedef __attribute__((ext_vector_type(8)))  float    v8f;

#define TN 4   // 16-wide N tiles per wave (16x64 C macro-tile)

// ---------------------------------------------------------------------------
// WMMA GEMM:  C[b][o][n] = [R[b][o][n] +] act( sum_k A[o][k]*BT[b][n][k] + bias[o] )
// A : f16 row-major O_pad x K (shared over batch), O_pad = ceil16(O)
// BT: f16 TRANSPOSED activations, Npad x K per batch (Npad = ceil64(N)); rows
//     beyond N hold garbage -> only feeds unsaved C columns.
// C : f32, row stride ldC, per-batch cStride.  R: optional residual (after relu).
// All fragment loads are unguarded 16B vector loads (two b128 per fragment).
// ---------------------------------------------------------------------------
__global__ __launch_bounds__(32)
void wmma_gemm_kernel(const _Float16* __restrict__ A, const _Float16* __restrict__ BT,
                      const float* __restrict__ bias, const float* __restrict__ Rres,
                      float* __restrict__ C, int O, int K, int Ncols, int ldC,
                      long bStride, long cStride, long rStride, int relu)
{
    const int lane   = threadIdx.x & 31;
    const int laneLo = lane & 15;
    const int laneHi = lane >> 4;
    const int n0     = blockIdx.x * (16 * TN);
    const int tileM  = blockIdx.y;
    const int batch  = blockIdx.z;

    const int arow = tileM * 16 + laneLo;
    // A: halves 0..7 at k0+laneHi*8, halves 8..15 at k0+16+laneHi*8
    const _Float16* Ap = A + (size_t)arow * K + (laneHi << 3);
    // B: halves 0..7 at k0+laneHi*8, halves 8..15 at k0+16+laneHi*8 of row n (transposed layout)
    const _Float16* Bp = BT + (size_t)batch * bStride + (laneHi << 3);

    v8f acc[TN];
#pragma unroll
    for (int t = 0; t < TN; ++t) acc[t] = (v8f){};

    for (int k0 = 0; k0 < K; k0 += 32) {
        v8h a0 = *(const v8h*)(Ap + k0);
        v8h a1 = *(const v8h*)(Ap + k0 + 16);
        v16h a = __builtin_shufflevector(a0, a1, 0,1,2,3,4,5,6,7,8,9,10,11,12,13,14,15);
#pragma unroll
        for (int t = 0; t < TN; ++t) {
            const _Float16* bp = Bp + (size_t)(n0 + t * 16 + laneLo) * K + k0;
            v8h b0 = *(const v8h*)(bp);
            v8h b1 = *(const v8h*)(bp + 16);
            v16h bf = __builtin_shufflevector(b0, b1, 0,1,2,3,4,5,6,7,8,9,10,11,12,13,14,15);
            acc[t] = __builtin_amdgcn_wmma_f32_16x16x32_f16(false, a, false, bf,
                                                            (short)0, acc[t], false, false);
        }
    }

    float* Cp = C + (size_t)batch * cStride;
    const float* Rp = Rres ? (Rres + (size_t)batch * rStride) : nullptr;
#pragma unroll
    for (int t = 0; t < TN; ++t) {
        int n = n0 + t * 16 + laneLo;
#pragma unroll
        for (int v = 0; v < 8; ++v) {
            int m = tileM * 16 + (laneHi << 3) + v;
            if (m < O && n < Ncols) {
                float r = acc[t][v];
                if (bias) r += bias[m];
                if (relu) r = fmaxf(r, 0.f);
                if (Rp)   r += Rp[(size_t)m * ldC + n];
                Cp[(size_t)m * ldC + n] = r;
            }
        }
    }
}

// ---------------------------------------------------------------------------
__global__ void cvt_f32_f16(const float* __restrict__ s, _Float16* __restrict__ d, size_t n)
{
    size_t i = (size_t)blockIdx.x * blockDim.x + threadIdx.x;
    if (i < n) d[i] = (_Float16)s[i];
}

// f32 [b][K][N] row-major  ->  f16 transposed [b][Npad][K]
__global__ void cvt_f32_f16_T(const float* __restrict__ s, _Float16* __restrict__ d,
                              int K, int N, int Npad, long sB, long dB, size_t total)
{
    size_t i = (size_t)blockIdx.x * blockDim.x + threadIdx.x;
    if (i >= total) return;
    int n = (int)(i % N);
    int k = (int)((i / N) % K);
    int b = (int)(i / ((size_t)N * K));
    d[(size_t)b * dB + (size_t)n * K + k] = (_Float16)s[(size_t)b * sB + (size_t)k * N + n];
}

__global__ void newp_kernel(const float* __restrict__ p, float* __restrict__ outp)
{
    int i = blockIdx.x * blockDim.x + threadIdx.x;
    if (i >= PB * PNP) return;
    int b = i / PNP, n = i % PNP;
    const float* s = p + ((size_t)b * PN + (size_t)n * 4) * 3;
    float* d = outp + (size_t)i * 3;
    d[0] = s[0]; d[1] = s[1]; d[2] = s[2];
}

// Ball query: one wave per query; collect first 32 in-radius indices in order.
__global__ void ball_query_kernel(const float* __restrict__ p,
                                  const float* __restrict__ newp,
                                  int* __restrict__ nbr)
{
    int gid  = blockIdx.x * blockDim.x + threadIdx.x;
    int wave = gid >> 5;
    int lane = threadIdx.x & 31;
    if (wave >= PB * PNP) return;
    int b = wave / PNP;
    const float* q = newp + (size_t)wave * 3;
    float qx = q[0], qy = q[1], qz = q[2];
    int* out = nbr + (size_t)wave * 32;
    int cnt = 0, first = PN - 1;
    for (int base = 0; base < PN && cnt < 32; base += 32) {
        int j = base + lane;
        const float* pp = p + ((size_t)b * PN + j) * 3;
        float dx = pp[0] - qx, dy = pp[1] - qy, dz = pp[2] - qz;
        bool ok = (dx * dx + dy * dy + dz * dz) <= PR2;   // no j < PN guard: PN=6000 is not a multiple of 32
        unsigned mask = __builtin_amdgcn_ballot_w32(ok);
        while (mask && cnt < 32) {
            int bit = __builtin_ctz(mask);
            mask &= mask - 1;
            int jj = base + bit;
            if (cnt == 0) first = jj;
            if (lane == 0) out[cnt] = jj;
            ++cnt;
        }
    }
    if (lane == 0)
        for (int c = cnt; c < 32; ++c) out[c] = first;
}

// fmax16 (transposed [b][n][o]) = max_s( pe_h + relu(Yall[o,nbr]-Yall[o,4n]+bsf) )
__global__ void combine_scale_kernel(const float* __restrict__ Yall,
                                     const int* __restrict__ nbr,
                                     const float* __restrict__ p,
                                     const float* __restrict__ newp,
                                     const float* __restrict__ Wsp_i,
                                     const float* __restrict__ bsp_i,
                                     const float* __restrict__ bsf_i,
                                     _Float16* __restrict__ fmax16,
                                     float* __restrict__ pe_out, int ns)
{
    int i = blockIdx.x * blockDim.x + threadIdx.x;
    if (i >= PB * PCH * PNP) return;
    int n = i % PNP;
    int o = (i / PNP) % PCH;
    int b = i / (PNP * PCH);

    const float w0 = Wsp_i[o * 3 + 0], w1 = Wsp_i[o * 3 + 1], w2 = Wsp_i[o * 3 + 2];
    const float bs = bsp_i[o], bf = bsf_i[o];
    const float* q = newp + ((size_t)b * PNP + n) * 3;
    const float qx = q[0], qy = q[1], qz = q[2];
    const float* Yb = Yall + ((size_t)b * PCH + o) * PN;
    const float  z  = Yb[(size_t)n * 4];
    const int* nb = nbr + ((size_t)b * PNP + n) * 32;

    float mx = -3.0e38f;
    for (int s = 0; s < ns; ++s) {
        int j = nb[s];
        const float* pj = p + ((size_t)b * PN + j) * 3;
        float pe = fmaxf(w0 * (pj[0] - qx) + w1 * (pj[1] - qy) + w2 * (pj[2] - qz) + bs, 0.f);
        float v  = pe + fmaxf(Yb[j] - z + bf, 0.f);
        mx = fmaxf(mx, v);
        if (pe_out) pe_out[((size_t)(b * PCH + o) * PNP + n) * 32 + s] = pe;
    }
    fmax16[((size_t)b * PNPPAD + n) * PCH + o] = (_Float16)mx;
}

// xin16 (transposed [b][n][c]) = localf + relu(Wgp . new_p + bgp)
__global__ void addgpe_kernel(const float* __restrict__ localf, const float* __restrict__ newp,
                              const float* __restrict__ Wgp, const float* __restrict__ bgp,
                              _Float16* __restrict__ xin16)
{
    size_t i = (size_t)blockIdx.x * blockDim.x + threadIdx.x;
    if (i >= (size_t)PB * PCOUT * PNP) return;
    int c = (int)(i % PCOUT);
    int n = (int)((i / PCOUT) % PNP);
    int b = (int)(i / ((size_t)PCOUT * PNP));
    const float* q = newp + ((size_t)b * PNP + n) * 3;
    float g = fmaxf(Wgp[c * 3] * q[0] + Wgp[c * 3 + 1] * q[1] + Wgp[c * 3 + 2] * q[2] + bgp[c], 0.f);
    xin16[((size_t)b * PNPPAD + n) * PCOUT + c] =
        (_Float16)(localf[((size_t)b * PCOUT + c) * PNP + n] + g);
}

__global__ void slot_softmax_lp_kernel(const float* __restrict__ logits,
                                       const float* __restrict__ newp,
                                       float* __restrict__ LP)
{
    __shared__ float sred[256];
    int bm = blockIdx.x;
    int b  = bm / PM;
    int t  = threadIdx.x;
    const float* row = logits + (size_t)bm * PNP;

    float mx = -3.0e38f;
    for (int n = t; n < PNP; n += 256) mx = fmaxf(mx, row[n]);
    sred[t] = mx; __syncthreads();
    for (int s = 128; s > 0; s >>= 1) { if (t < s) sred[t] = fmaxf(sred[t], sred[t + s]); __syncthreads(); }
    mx = sred[0]; __syncthreads();

    float sum = 0.f;
    for (int n = t; n < PNP; n += 256) sum += expf(row[n] - mx);
    sred[t] = sum; __syncthreads();
    for (int s = 128; s > 0; s >>= 1) { if (t < s) sred[t] += sred[t + s]; __syncthreads(); }
    float inv = 1.f / sred[0]; __syncthreads();

    float px = 0.f, py = 0.f, pz = 0.f;
    for (int n = t; n < PNP; n += 256) {
        float pr = expf(row[n] - mx) * inv;
        const float* q = newp + ((size_t)b * PNP + n) * 3;
        px += pr * q[0]; py += pr * q[1]; pz += pr * q[2];
    }
    sred[t] = px; __syncthreads();
    for (int s = 128; s > 0; s >>= 1) { if (t < s) sred[t] += sred[t + s]; __syncthreads(); }
    px = sred[0]; __syncthreads();
    sred[t] = py; __syncthreads();
    for (int s = 128; s > 0; s >>= 1) { if (t < s) sred[t] += sred[t + s]; __syncthreads(); }
    py = sred[0]; __syncthreads();
    sred[t] = pz; __syncthreads();
    for (int s = 128; s > 0; s >>= 1) { if (t < s) sred[t] += sred[t + s]; __syncthreads(); }
    pz = sred[0];
    if (t == 0) { LP[(size_t)bm * 3 + 0] = px; LP[(size_t)bm * 3 + 1] = py; LP[(size_t)bm * 3 + 2] = pz; }
}

__global__ void argmin_j1_kernel(const float* __restrict__ LP, const float* __restrict__ newp,
                                 int* __restrict__ j1)
{
    int i = blockIdx.x * blockDim.x + threadIdx.x;
    if (i >= PB * PM) return;
    int b = i / PM;
    float lx = LP[i * 3], ly = LP[i * 3 + 1], lz = LP[i * 3 + 2];
    float best = 3.0e38f; int bi = 0;
    for (int n = 0; n < PNP; ++n) {
        const float* q = newp + ((size_t)b * PNP + n) * 3;
        float dx = lx - q[0], dy = ly - q[1], dz = lz - q[2];
        float d = dx * dx + dy * dy + dz * dz;
        if (d < best) { best = d; bi = n; }
    }
    j1[i] = bi;
}

__global__ void argmin_j2_kernel(const float* __restrict__ newp, const float* __restrict__ LP,
                                 int* __restrict__ j2)
{
    int i = blockIdx.x * blockDim.x + threadIdx.x;
    if (i >= PB * PNP) return;
    int b = i / PNP;
    const float* q = newp + (size_t)i * 3;
    float best = 3.0e38f; int bi = 0;
    for (int m = 0; m < PM; ++m) {
        const float* l = LP + ((size_t)b * PM + m) * 3;
        float dx = q[0] - l[0], dy = q[1] - l[1], dz = q[2] - l[2];
        float d = dx * dx + dy * dy + dz * dz;
        if (d < best) { best = d; bi = m; }
    }
    j2[i] = bi;
}

// lpf16 (transposed [b][m][c]) = localf[b][c][j1[b][m]]
__global__ void gather_lpf_kernel(const float* __restrict__ localf, const int* __restrict__ j1,
                                  _Float16* __restrict__ lpf16)
{
    int i = blockIdx.x * blockDim.x + threadIdx.x;
    if (i >= PB * PCOUT * PM) return;
    int c = i % PCOUT;
    int m = (i / PCOUT) % PM;
    int b = i / (PCOUT * PM);
    lpf16[((size_t)b * PMPAD + m) * PCOUT + c] =
        (_Float16)localf[((size_t)b * PCOUT + c) * PNP + j1[b * PM + m]];
}

__global__ void attn_softmax_kernel(const float* __restrict__ q, const float* __restrict__ k,
                                    float* __restrict__ attn)
{
    int i = blockIdx.x * blockDim.x + threadIdx.x;
    if (i >= PB * PH * PM) return;
    int qm = i % PM;
    int h  = (i / PM) % PH;
    int b  = i / (PM * PH);
    const float* qp = q + ((size_t)b * PCOUT + h * 64) * PM + qm;
    const float* kb = k + ((size_t)b * PCOUT + h * 64) * PM;
    float* arow = attn + ((size_t)(b * PH + h) * PM + qm) * PM;
    float mx = -3.0e38f;
    for (int kk = 0; kk < PM; ++kk) {
        float s = 0.f;
        for (int d = 0; d < 64; ++d) s += qp[(size_t)d * PM] * kb[(size_t)d * PM + kk];
        s *= 0.125f;
        arow[kk] = s;
        mx = fmaxf(mx, s);
    }
    float sum = 0.f;
    for (int kk = 0; kk < PM; ++kk) { float e = expf(arow[kk] - mx); arow[kk] = e; sum += e; }
    float inv = 1.f / sum;
    for (int kk = 0; kk < PM; ++kk) arow[kk] *= inv;
}

__global__ void attn_vals_kernel(const float* __restrict__ attn, const float* __restrict__ v,
                                 float* __restrict__ vals)
{
    int i = blockIdx.x * blockDim.x + threadIdx.x;
    if (i >= PB * PCOUT * PM) return;
    int qm = i % PM;
    int o  = (i / PM) % PCOUT;
    int b  = i / (PM * PCOUT);
    int h  = o >> 6;
    const float* arow = attn + ((size_t)(b * PH + h) * PM + qm) * PM;
    const float* vrow = v + ((size_t)b * PCOUT + o) * PM;
    float s = 0.f;
    for (int kk = 0; kk < PM; ++kk) s += arow[kk] * vrow[kk];
    vals[i] = s;
}

// gf0 (f32 row-major) + gf016 (transposed) = LPgf gathered by j2
__global__ void gather_gf_kernel(const float* __restrict__ LPgf, const int* __restrict__ j2,
                                 float* __restrict__ gf0, _Float16* __restrict__ gf016)
{
    size_t i = (size_t)blockIdx.x * blockDim.x + threadIdx.x;
    if (i >= (size_t)PB * PCOUT * PNP) return;
    int c = (int)(i % PCOUT);
    int n = (int)((i / PCOUT) % PNP);
    int b = (int)(i / ((size_t)PCOUT * PNP));
    float v = LPgf[((size_t)b * PCOUT + c) * PM + j2[b * PNP + n]];
    gf0[((size_t)b * PCOUT + c) * PNP + n] = v;
    gf016[((size_t)b * PNPPAD + n) * PCOUT + c] = (_Float16)v;
}

__global__ void blend_kernel(const float* __restrict__ localf, const float* __restrict__ gf,
                             const float* __restrict__ alpha,
                             float* __restrict__ fout0, _Float16* __restrict__ fout016)
{
    size_t i = (size_t)blockIdx.x * blockDim.x + threadIdx.x;
    if (i >= (size_t)PB * PCOUT * PNP) return;
    int n = (int)(i % PNP);
    int c = (int)((i / PNP) % PCOUT);
    int b = (int)(i / ((size_t)PNP * PCOUT));
    float a = 1.f / (1.f + expf(-alpha[0]));
    float v = localf[i] * a + gf[i] * (1.f - a);
    fout0[i] = v;
    fout016[((size_t)b * PNPPAD + n) * PCOUT + c] = (_Float16)v;
}

// ---------------------------------------------------------------------------
extern "C" void kernel_launch(void* const* d_in, const int* in_sizes, int n_in,
                              void* d_out, int out_size, void* d_ws, size_t ws_size,
                              hipStream_t stream)
{
    const float* p     = (const float*)d_in[0];
    const float* f     = (const float*)d_in[1];
    const float* W_pre = (const float*)d_in[3];
    const float* b_pre = (const float*)d_in[4];
    const float* Wsp   = (const float*)d_in[5];
    const float* bsp   = (const float*)d_in[6];
    const float* Wsf   = (const float*)d_in[7];
    const float* bsf   = (const float*)d_in[8];
    const float* Wmlp  = (const float*)d_in[9];
    const float* bmlp  = (const float*)d_in[10];
    const float* Wfin  = (const float*)d_in[11];
    const float* bfin  = (const float*)d_in[12];
    const float* Wgp   = (const float*)d_in[13];
    const float* bgp   = (const float*)d_in[14];
    const float* Wg1   = (const float*)d_in[15];
    const float* bg1   = (const float*)d_in[16];
    const float* Wslot = (const float*)d_in[17];
    const float* Wq    = (const float*)d_in[18];
    const float* bq    = (const float*)d_in[19];
    const float* Wk    = (const float*)d_in[20];
    const float* bk    = (const float*)d_in[21];
    const float* Wv    = (const float*)d_in[22];
    const float* bv    = (const float*)d_in[23];
    const float* Wo    = (const float*)d_in[24];
    const float* bo    = (const float*)d_in[25];
    const float* W2    = (const float*)d_in[26];
    const float* b2    = (const float*)d_in[27];
    const float* W3    = (const float*)d_in[28];
    const float* b3    = (const float*)d_in[29];
    const float* alpha = (const float*)d_in[30];

    float* outp  = (float*)d_out;
    float* newp  = outp;                                       // (B,NP,3)
    float* fout  = outp + (size_t)PB * PNP * 3;                // (B,256,1500)
    float* peout = fout + (size_t)PB * PCOUT * PNP;            // (B,64,1500,32)

    char* wp = (char*)d_ws;
    auto alloc = [&](size_t bytes) -> void* {
        void* r = (void*)wp;
        wp += (bytes + 255) & ~(size_t)255;
        return r;
    };
    _Float16* wPre16  = (_Float16*)alloc((size_t)PCH * PCIN * 2);
    _Float16* wSf16   = (_Float16*)alloc((size_t)PH * PCH * PCH * 2);
    _Float16* wMlp16  = (_Float16*)alloc((size_t)PH * PCH * PCH * 2);
    _Float16* wFin16  = (_Float16*)alloc((size_t)PCOUT * PCOUT * 2);
    _Float16* wG116   = (_Float16*)alloc((size_t)PCOUT * PCOUT * 2);
    _Float16* wSlot16 = (_Float16*)alloc((size_t)160 * PCOUT * 2);       // 150 -> 160 rows
    _Float16* wQ16    = (_Float16*)alloc((size_t)PCOUT * PCOUT * 2);
    _Float16* wK16    = (_Float16*)alloc((size_t)PCOUT * PCOUT * 2);
    _Float16* wV16    = (_Float16*)alloc((size_t)PCOUT * PCOUT * 2);
    _Float16* wO16    = (_Float16*)alloc((size_t)PCOUT * PCOUT * 2);
    _Float16* w216    = (_Float16*)alloc((size_t)PCOUT * PCOUT * 2);
    _Float16* w316    = (_Float16*)alloc((size_t)PCOUT * PCOUT * 2);
    _Float16* f16in   = (_Float16*)alloc((size_t)PB * PNPAD * PCIN * 2);   // T
    float*    f0      = (float*)   alloc((size_t)PB * PCH * PN * 4);
    _Float16* f016    = (_Float16*)alloc((size_t)PB * PNPAD * PCH * 2);    // T
    int*      nbr     = (int*)     alloc((size_t)PB * PNP * 32 * 4);
    float*    Yall    = (float*)   alloc((size_t)PB * PCH * PN * 4);
    _Float16* fmax16  = (_Float16*)alloc((size_t)PB * PNPPAD * PCH * 2);   // T
    float*    fcat    = (float*)   alloc((size_t)PB * PCOUT * PNP * 4);
    _Float16* fcat16  = (_Float16*)alloc((size_t)PB * PNPPAD * PCOUT * 2); // T
    float*    localf  = (float*)   alloc((size_t)PB * PCOUT * PNP * 4);
    _Float16* xin16   = (_Float16*)alloc((size_t)PB * PNPPAD * PCOUT * 2); // T
    float*    xbuf    = (float*)   alloc((size_t)PB * PCOUT * PNP * 4);
    _Float16* x16     = (_Float16*)alloc((size_t)PB * PNPPAD * PCOUT * 2); // T
    float*    logits  = (float*)   alloc((size_t)PB * PM * PNP * 4);
    float*    LP      = (float*)   alloc((size_t)PB * PM * 3 * 4);
    int*      j1      = (int*)     alloc((size_t)PB * PM * 4);
    int*      j2      = (int*)     alloc((size_t)PB * PNP * 4);
    _Float16* lpf16   = (_Float16*)alloc((size_t)PB * PMPAD * PCOUT * 2);  // T
    float*    qb      = (float*)   alloc((size_t)PB * PCOUT * PM * 4);
    float*    kb      = (float*)   alloc((size_t)PB * PCOUT * PM * 4);
    float*    vb      = (float*)   alloc((size_t)PB * PCOUT * PM * 4);
    float*    attn    = (float*)   alloc((size_t)PB * PH * PM * PM * 4);
    float*    valsb   = (float*)   alloc((size_t)PB * PCOUT * PM * 4);
    _Float16* vals16  = (_Float16*)alloc((size_t)PB * PMPAD * PCOUT * 2);  // T
    float*    LPgf    = (float*)   alloc((size_t)PB * PCOUT * PM * 4);
    float*    gf0     = (float*)   alloc((size_t)PB * PCOUT * PNP * 4);
    _Float16* gf016   = (_Float16*)alloc((size_t)PB * PNPPAD * PCOUT * 2); // T
    float*    gfb     = (float*)   alloc((size_t)PB * PCOUT * PNP * 4);
    float*    fout0   = (float*)   alloc((size_t)PB * PCOUT * PNP * 4);
    _Float16* fout016 = (_Float16*)alloc((size_t)PB * PNPPAD * PCOUT * 2); // T

    auto cvt = [&](const float* s, _Float16* d, size_t n) {
        cvt_f32_f16<<<(unsigned)((n + 255) / 256), 256, 0, stream>>>(s, d, n);
    };
    auto cvtT = [&](const float* s, _Float16* d, int K, int N, int Npad) {
        size_t total = (size_t)PB * K * N;
        cvt_f32_f16_T<<<(unsigned)((total + 255) / 256), 256, 0, stream>>>(
            s, d, K, N, Npad, (long)K * N, (long)Npad * K, total);
    };
    auto gemm = [&](const _Float16* A, const _Float16* BT, const float* bias, const float* R,
                    float* C, int O, int K, int N, int Npad, int ldC,
                    long cS, long rS, int relu) {
        dim3 g((N + 63) / 64, (O + 15) / 16, PB);
        wmma_gemm_kernel<<<g, 32, 0, stream>>>(A, BT, bias, R, C, O, K, N, ldC,
                                               (long)Npad * K, cS, rS, relu);
    };

    // ---- weight conversions (row-major f16) ----
    cvt(W_pre, wPre16, (size_t)PCH * PCIN);
    cvt(Wsf,   wSf16,  (size_t)PH * PCH * PCH);
    cvt(Wmlp,  wMlp16, (size_t)PH * PCH * PCH);
    cvt(Wfin,  wFin16, (size_t)PCOUT * PCOUT);
    cvt(Wg1,   wG116,  (size_t)PCOUT * PCOUT);
    cvt(Wslot, wSlot16,(size_t)PM * PCOUT);
    cvt(Wq, wQ16, (size_t)PCOUT * PCOUT);
    cvt(Wk, wK16, (size_t)PCOUT * PCOUT);
    cvt(Wv, wV16, (size_t)PCOUT * PCOUT);
    cvt(Wo, wO16, (size_t)PCOUT * PCOUT);
    cvt(W2, w216, (size_t)PCOUT * PCOUT);
    cvt(W3, w316, (size_t)PCOUT * PCOUT);

    // ---- new_p ----
    newp_kernel<<<(PB * PNP + 255) / 256, 256, 0, stream>>>(p, newp);

    // ---- f0 = relu(W_pre @ f + b_pre) ----
    cvtT(f, f16in, PCIN, PN, PNPAD);
    gemm(wPre16, f16in, b_pre, nullptr, f0, PCH, PCIN, PN, PNPAD, PN,
         (long)PCH * PN, 0, 1);
    cvtT(f0, f016, PCH, PN, PNPAD);

    // ---- ball query ----
    ball_query_kernel<<<(PB * PNP * 32 + 255) / 256, 256, 0, stream>>>(p, newp, nbr);

    // ---- per-scale neighbor aggregation ----
    const int nsamples[4] = {4, 8, 16, 32};
    for (int i = 0; i < 4; ++i) {
        gemm(wSf16 + (size_t)i * PCH * PCH, f016, nullptr, nullptr, Yall,
             PCH, PCH, PN, PNPAD, PN, (long)PCH * PN, 0, 0);
        combine_scale_kernel<<<(PB * PCH * PNP + 255) / 256, 256, 0, stream>>>(
            Yall, nbr, p, newp, Wsp + (size_t)i * PCH * 3, bsp + (size_t)i * PCH,
            bsf + (size_t)i * PCH, fmax16, (i == 3) ? peout : nullptr, nsamples[i]);
        gemm(wMlp16 + (size_t)i * PCH * PCH, fmax16, bmlp + (size_t)i * PCH, nullptr,
             fcat + (size_t)i * PCH * PNP, PCH, PCH, PNP, PNPPAD, PNP,
             (long)PCOUT * PNP, 0, 1);
    }

    // ---- local_f = fcat + relu(Wfin @ fcat + bfin) ----
    cvtT(fcat, fcat16, PCOUT, PNP, PNPPAD);
    gemm(wFin16, fcat16, bfin, fcat, localf, PCOUT, PCOUT, PNP, PNPPAD, PNP,
         (long)PCOUT * PNP, (long)PCOUT * PNP, 1);

    // ---- x = relu(Wg1 @ (local_f + gpe) + bg1) ----
    addgpe_kernel<<<(unsigned)(((size_t)PB * PCOUT * PNP + 255) / 256), 256, 0, stream>>>(
        localf, newp, Wgp, bgp, xin16);
    gemm(wG116, xin16, bg1, nullptr, xbuf, PCOUT, PCOUT, PNP, PNPPAD, PNP,
         (long)PCOUT * PNP, 0, 1);
    cvtT(xbuf, x16, PCOUT, PNP, PNPPAD);

    // ---- slots ----
    gemm(wSlot16, x16, nullptr, nullptr, logits, PM, PCOUT, PNP, PNPPAD, PNP,
         (long)PM * PNP, 0, 0);
    slot_softmax_lp_kernel<<<PB * PM, 256, 0, stream>>>(logits, newp, LP);

    // ---- j1, LP_f gather ----
    argmin_j1_kernel<<<(PB * PM + 255) / 256, 256, 0, stream>>>(LP, newp, j1);
    gather_lpf_kernel<<<(PB * PCOUT * PM + 255) / 256, 256, 0, stream>>>(localf, j1, lpf16);

    // ---- attention ----
    gemm(wQ16, lpf16, bq, nullptr, qb, PCOUT, PCOUT, PM, PMPAD, PM, (long)PCOUT * PM, 0, 0);
    gemm(wK16, lpf16, bk, nullptr, kb, PCOUT, PCOUT, PM, PMPAD, PM, (long)PCOUT * PM, 0, 0);
    gemm(wV16, lpf16, bv, nullptr, vb, PCOUT, PCOUT, PM, PMPAD, PM, (long)PCOUT * PM, 0, 0);
    attn_softmax_kernel<<<(PB * PH * PM + 255) / 256, 256, 0, stream>>>(qb, kb, attn);
    attn_vals_kernel<<<(PB * PCOUT * PM + 255) / 256, 256, 0, stream>>>(attn, vb, valsb);
    cvtT(valsb, vals16, PCOUT, PM, PMPAD);
    gemm(wO16, vals16, bo, nullptr, LPgf, PCOUT, PCOUT, PM, PMPAD, PM, (long)PCOUT * PM, 0, 0);

    // ---- j2, gf gather, gf = gf + relu(W2 @ gf + b2) ----
    argmin_j2_kernel<<<(PB * PNP + 255) / 256, 256, 0, stream>>>(newp, LP, j2);
    gather_gf_kernel<<<(unsigned)(((size_t)PB * PCOUT * PNP + 255) / 256), 256, 0, stream>>>(
        LPgf, j2, gf0, gf016);
    gemm(w216, gf016, b2, gf0, gfb, PCOUT, PCOUT, PNP, PNPPAD, PNP,
         (long)PCOUT * PNP, (long)PCOUT * PNP, 1);

    // ---- blend + final residual MLP -> fout ----
    blend_kernel<<<(unsigned)(((size_t)PB * PCOUT * PNP + 255) / 256), 256, 0, stream>>>(
        localf, gfb, alpha, fout0, fout016);
    gemm(w316, fout016, b3, fout0, fout, PCOUT, PCOUT, PNP, PNPPAD, PNP,
         (long)PCOUT * PNP, (long)PCOUT * PNP, 1);
}
